// SynapticMemoryCell_70068096467276
// MI455X (gfx1250) — compile-verified, matrix operand order checked by reading, not run on hardware
//
#include <hip/hip_runtime.h>
#include <hip/hip_bf16.h>

// ---- problem constants (from reference) ----
#define BQ    4096
#define CAP   32768
#define DK    512
#define DV    512
#define TOPK  32
#define MTILE 64                        // query rows per block
#define NGRP  4                         // column groups (4 waves each)
#define CTPG4 (CAP / 16 / NGRP)         // 16-key tiles per group = 512
#define NEG_BIG (-1.0e30f)

typedef __attribute__((ext_vector_type(16))) __bf16 v16bf;
typedef __attribute__((ext_vector_type(8)))  __bf16 v8bf;
typedef __attribute__((ext_vector_type(8)))  float  v8f;

union FragBF {
    v16bf v;
    v8bf  h[2];
};

// -------------------------------------------------------------------------
// Kernel 1: row-wise L2 normalize (f32 -> bf16). One wave per row.
// -------------------------------------------------------------------------
__global__ void norm_bf16_kernel(const float* __restrict__ src,
                                 __bf16* __restrict__ dst, int rows) {
    const int w    = threadIdx.x >> 5;
    const int lane = threadIdx.x & 31;
    const int row  = blockIdx.x * 8 + w;
    if (row >= rows) return;

    const float* r = src + (size_t)row * DK;
    float v[16];
    float ss = 0.0f;
#pragma unroll
    for (int i = 0; i < 16; ++i) {
        v[i] = r[i * 32 + lane];
        ss += v[i] * v[i];
    }
#pragma unroll
    for (int off = 16; off > 0; off >>= 1) ss += __shfl_xor(ss, off);

    const float inv = 1.0f / fmaxf(sqrtf(ss), 1e-12f);
    __bf16* o = dst + (size_t)row * DK;
#pragma unroll
    for (int i = 0; i < 16; ++i) o[i * 32 + lane] = (__bf16)(v[i] * inv);
}

// -------------------------------------------------------------------------
// Kernel 2: fused sims-GEMM (WMMA bf16, triple-buffered async-LDS keys)
// Grid: BQ/64 = 64 blocks, 512 threads (16 waves = 4 waves/SIMD for
// LDS-latency hiding). Wave w: M-subtile (w&3), column group (w>>2).
// Block reads kn exactly once => 2 GB total L2 traffic.
// -------------------------------------------------------------------------
__global__ void __launch_bounds__(512)
fused_topk_kernel(const __bf16* __restrict__ qn,     // [BQ, DK]  normalized bf16
                  const __bf16* __restrict__ knb,    // [CAP, DK] normalized bf16
                  const float*  __restrict__ strengths,
                  float* __restrict__ outScores,     // [BQ, TOPK]
                  int*   __restrict__ outIdx) {      // [BQ, TOPK]
    __shared__ __bf16 sB[NGRP][3][16 * DK];  // 192 KB triple-buffered staging
    __shared__ float  sScore[16][256];       // 16 KB per-wave 16x16 score tile
    __shared__ float  sTopV[16][16][TOPK];   // 32 KB per-wave top lists
    __shared__ int    sTopI[16][16][TOPK];   // 32 KB

    const int w    = threadIdx.x >> 5;   // 0..15
    const int lane = threadIdx.x & 31;
    const int msub = w & 3;              // M-subtile (16 rows)
    const int g    = w >> 2;             // column group 0..3
    const int nrow = lane & 15;          // M (A) / N (B) lane index
    const int half = lane >> 4;          // K-half per 16-bit WMMA layout

    // ---- hoist A fragments into registers (constant over column sweep) ----
    FragBF afrag[16];
    {
        const __bf16* aRow =
            qn + ((size_t)blockIdx.x * MTILE + msub * 16 + nrow) * DK + half * 8;
#pragma unroll
        for (int ks = 0; ks < 16; ++ks) {
            afrag[ks].h[0] = *(const v8bf*)(aRow + ks * 32);
            afrag[ks].h[1] = *(const v8bf*)(aRow + ks * 32 + 16);
        }
    }
    if (lane < 16) {
        for (int s = 0; s < TOPK; ++s) {
            sTopV[w][lane][s] = NEG_BIG;
            sTopI[w][lane][s] = 0;
        }
    }

    // ---- async staging of one 16-row key tile (16 KB) per group ----
    const int gl = msub * 32 + lane;     // lane id within group: 0..127
    auto stage = [&](int buf, int tile) {
        // 16 rows contiguous: 16 * DK bf16 = 16 KB = 1024 x 16B chunks
        const uint64_t gbase =
            (uint64_t)(uintptr_t)(knb + (size_t)(g * CTPG4 + tile) * 16 * DK);
        const unsigned lbase = (unsigned)(uintptr_t)(&sB[g][buf][0]);
#pragma unroll
        for (int j = 0; j < 8; ++j) {
            const int chunk = gl + 128 * j;              // 0..1023
            asm volatile("global_load_async_to_lds_b128 %0, %1, off"
                         :: "v"((unsigned)(lbase + chunk * 16)),
                            "v"(gbase + (uint64_t)chunk * 16)
                         : "memory");
        }
    };

    float curMin = NEG_BIG;
    int   minPos = 0;

    stage(0, 0);

    for (int t = 0; t < CTPG4; ++t) {
        const int cur = t % 3;
        if (t + 1 < CTPG4) {
            stage((t + 1) % 3, t + 1);
            // in-order async completion: <=8 outstanding forces tile t done
            asm volatile("s_wait_asynccnt 0x8" ::: "memory");
        } else {
            asm volatile("s_wait_asynccnt 0x0" ::: "memory");
        }
        __syncthreads();   // tile t visible group-wide; separates re-stage WAR

        const __bf16* bBase = &sB[g][cur][0] + (size_t)nrow * DK + half * 8;
        // depth-4 rotating B-fragment pipeline over the 16 k-steps
        FragBF bf[4];
#pragma unroll
        for (int p = 0; p < 4; ++p) {
            bf[p].h[0] = *(const v8bf*)(bBase + p * 32);
            bf[p].h[1] = *(const v8bf*)(bBase + p * 32 + 16);
        }
        v8f acc = {};
#pragma unroll
        for (int ks = 0; ks < 16; ++ks) {
            if (ks + 4 < 16) {
                bf[(ks + 4) & 3].h[0] = *(const v8bf*)(bBase + (ks + 4) * 32);
                bf[(ks + 4) & 3].h[1] = *(const v8bf*)(bBase + (ks + 4) * 32 + 16);
            }
            acc = __builtin_amdgcn_wmma_f32_16x16x32_bf16(
                false, afrag[ks].v, false, bf[ks & 3].v,
                (short)0, acc, false, false);
        }

        // dump 16x16 tile: VGPR r -> M = r + 8*half, N = lane&15
#pragma unroll
        for (int r = 0; r < 8; ++r)
            sScore[w][(r + half * 8) * 16 + nrow] = acc[r];
        asm volatile("" ::: "memory");

        // per-row replace-min top-32 update (lane == row, lanes 0..15)
        if (lane < 16) {
            const int row = lane;
            const int colBase = (g * CTPG4 + t) * 16;
            for (int c = 0; c < 16; ++c) {
                const float v = sScore[w][row * 16 + c];
                if (v > curMin) {
                    sTopV[w][row][minPos] = v;
                    sTopI[w][row][minPos] = colBase + c;
                    float m = sTopV[w][row][0];
                    int mp = 0;
                    for (int s = 1; s < TOPK; ++s) {
                        const float tv = sTopV[w][row][s];
                        if (tv < m) { m = tv; mp = s; }
                    }
                    curMin = m;
                    minPos = mp;
                }
            }
        }
        asm volatile("" ::: "memory");
    }
    __syncthreads();   // lists complete before cross-wave merge

    // ---- merge the 4 column-group lists per row, softmax, emit ----
    for (int rr = 0; rr < 4; ++rr) {
        const int row = w * 4 + rr;          // 0..63
        const int ms  = row >> 4;
        const int r16 = row & 15;
        float cv[NGRP];
        int   ci[NGRP];
#pragma unroll
        for (int tgt = 0; tgt < NGRP; ++tgt) {
            cv[tgt] = sTopV[ms + 4 * tgt][r16][lane];
            ci[tgt] = sTopI[ms + 4 * tgt][r16][lane];
        }

        float myVal = NEG_BIG;
        int   myIdx = 0;
        for (int k = 0; k < TOPK; ++k) {
            float lm = cv[0];
            int   ls = 0;
#pragma unroll
            for (int tgt = 1; tgt < NGRP; ++tgt)
                if (cv[tgt] > lm) { lm = cv[tgt]; ls = tgt; }
            float wm = lm;
#pragma unroll
            for (int off = 16; off > 0; off >>= 1) wm = fmaxf(wm, __shfl_xor(wm, off));
            const unsigned long long bal = __ballot(lm == wm);
            const int owner = (int)__builtin_ctzll(bal);
            const int cand  = ci[ls];
            const int idx   = __shfl(cand, owner);
            if (lane == owner) cv[ls] = NEG_BIG;
            if (lane == k) { myVal = wm; myIdx = idx; }
        }
        // weighted = sim * strength (TEMP == 1); softmax over 32 lanes
        const float wv = myVal * strengths[myIdx];
        float mx = wv;
#pragma unroll
        for (int off = 16; off > 0; off >>= 1) mx = fmaxf(mx, __shfl_xor(mx, off));
        const float e = __expf(wv - mx);
        float sum = e;
#pragma unroll
        for (int off = 16; off > 0; off >>= 1) sum += __shfl_xor(sum, off);

        const size_t gRow = (size_t)blockIdx.x * MTILE + row;
        outScores[gRow * TOPK + lane] = e / sum;
        outIdx[gRow * TOPK + lane]    = myIdx;
    }
}

// -------------------------------------------------------------------------
// Kernel 3: gather retrieved keys & values. 4096 blocks of pure
// b128-coalesced copies to saturate the 23.3 TB/s HBM on the output.
// -------------------------------------------------------------------------
__global__ void __launch_bounds__(256)
gather_kernel(const float* __restrict__ memK, const float* __restrict__ memV,
              const int* __restrict__ idxArr, float* __restrict__ out) {
    const int row = blockIdx.x;
    __shared__ int sIdx[TOPK];
    if (threadIdx.x < TOPK) sIdx[threadIdx.x] = idxArr[(size_t)row * TOPK + threadIdx.x];
    __syncthreads();

    float* outK = out + (size_t)row * TOPK * DK;
    float* outV = out + (size_t)BQ * TOPK * DK + (size_t)row * TOPK * DV;
    for (int i = threadIdx.x; i < TOPK * (DK / 4); i += 256) {
        const int e = i >> 7;        // entry
        const int c = i & 127;       // float4 chunk
        const int idx = sIdx[e];
        const float4 kv = *(const float4*)(memK + (size_t)idx * DK + c * 4);
        *(float4*)(outK + (size_t)e * DK + (size_t)c * 4) = kv;
        const float4 vv = *(const float4*)(memV + (size_t)idx * DV + c * 4);
        *(float4*)(outV + (size_t)e * DV + (size_t)c * 4) = vv;
    }
}

// -------------------------------------------------------------------------
extern "C" void kernel_launch(void* const* d_in, const int* in_sizes, int n_in,
                              void* d_out, int out_size, void* d_ws, size_t ws_size,
                              hipStream_t stream) {
    (void)in_sizes; (void)n_in; (void)out_size; (void)ws_size;
    const float* q  = (const float*)d_in[0];
    const float* mk = (const float*)d_in[1];
    const float* mv = (const float*)d_in[2];
    const float* ms = (const float*)d_in[3];

    __bf16* qn  = (__bf16*)d_ws;                         // 4 MB
    __bf16* kn  = qn + (size_t)BQ * DK;                  // 32 MB
    int*    idx = (int*)(kn + (size_t)CAP * DK);         // 512 KB

    float* out       = (float*)d_out;
    float* outScores = out + (size_t)2 * BQ * TOPK * DK;

    norm_bf16_kernel<<<BQ / 8, 256, 0, stream>>>(q, qn, BQ);
    norm_bf16_kernel<<<CAP / 8, 256, 0, stream>>>(mk, kn, CAP);
    fused_topk_kernel<<<BQ / MTILE, 512, 0, stream>>>(qn, kn, ms, outScores, idx);
    gather_kernel<<<BQ, 256, 0, stream>>>(mk, mv, idx, out);
}
